// VmambaMixer_48747878810259
// MI455X (gfx1250) — matrix operand order repaired and checked by reading, not compiled, not run on hardware
//
#include <hip/hip_runtime.h>
#include <hip/hip_bf16.h>
#include <math.h>

// ---------------- problem constants ----------------
#define BB   8
#define HH   56
#define WW   56
#define DM   96
#define DIN  192          // EXPAND*DM
#define DST  32           // D_STATE
#define RK   6            // DT_RANK
#define HD   32           // HEAD_D
#define KD   4            // directions
#define LL   (HH*WW)      // 3136
#define NT   (BB*LL)      // 25088 tokens
#define CPJ  70           // DT_RANK + 2*D_STATE
#define CPJP 80           // padded to 5 n-tiles

typedef __attribute__((ext_vector_type(16))) _Float16 v16h;
typedef __attribute__((ext_vector_type(8)))  float    v8f;

// ---------------- workspace layout (bytes, 256-aligned) ----------------
static constexpr size_t aln(size_t x){ return (x + 255) & ~size_t(255); }
static constexpr size_t OFF_XB   = 0;                                             // f16 x       [NT][96]
static constexpr size_t OFF_WI   = aln(OFF_XB   + (size_t)NT*DM*2);               // f16 inproj  [384][96]
static constexpr size_t OFF_WX   = aln(OFF_WI   + (size_t)2*DIN*DM*2);            // f16 xproj   [4][80][192] (zero-padded)
static constexpr size_t OFF_WO   = aln(OFF_WX   + (size_t)KD*CPJP*DIN*2);         // f16 outproj [96][192]
static constexpr size_t OFF_XZ   = aln(OFF_WO   + (size_t)DM*DIN*2);              // f32 xz      [NT][384]
static constexpr size_t OFF_XIB  = aln(OFF_XZ   + (size_t)NT*2*DIN*4);            // f16 silu(conv(xi)) [NT][192]
static constexpr size_t OFF_XDBL = aln(OFF_XIB  + (size_t)NT*DIN*2);              // f32 x_dbl   [4][8][3136][80]
static constexpr size_t OFF_DTDA = aln(OFF_XDBL + (size_t)KD*BB*LL*CPJP*4);       // f32x2 dt,dA [4][8][3136][6][2]
static constexpr size_t OFF_YS   = aln(OFF_DTDA + (size_t)KD*BB*LL*RK*8);         // f32 ys      [4*8][192][3136]
static constexpr size_t OFF_YB   = aln(OFF_YS   + (size_t)KD*BB*DIN*LL*4);        // f16 gelu(ln)*silu(z) [NT][192]

// scan-order -> spatial index for direction k
__device__ __forceinline__ int permk(int k, int l){
  switch(k){
    case 0:  return l;
    case 1:  { int hi = l % HH, wi = l / HH; return hi*WW + wi; }
    case 2:  return LL-1-l;
    default: { int l2 = LL-1-l; int hi = l2 % HH, wi = l2 / HH; return hi*WW + wi; }
  }
}

__device__ __forceinline__ float siluf(float x){ return x / (1.f + expf(-x)); }
__device__ __forceinline__ float softplusf(float x){ return x > 20.f ? x : log1pf(expf(x)); }

// A fragment: 16-bit A 16x32 layout (lane-half K split)
__device__ __forceinline__ v16h load_afrag(const _Float16* __restrict__ pa, int half){
  v16h f;
#pragma unroll
  for (int e = 0; e < 8; ++e){ f[e] = pa[16*half + e]; f[8+e] = pa[16*half + 8 + e]; }
  return f;
}
// B fragment: 16-bit B 32x16 layout -> 16 contiguous K from W row n (TN gemm)
__device__ __forceinline__ v16h load_bfrag(const _Float16* __restrict__ pb){
  v16h f;
#pragma unroll
  for (int e = 0; e < 16; ++e) f[e] = pb[e];
  return f;
}

// ---------------- conversions ----------------
__global__ void cvt_f16_kernel(const float* __restrict__ src, _Float16* __restrict__ dst, size_t n){
  size_t i = (size_t)blockIdx.x*blockDim.x + threadIdx.x;
  size_t stride = (size_t)gridDim.x*blockDim.x;
  for (; i < n; i += stride) dst[i] = (_Float16)src[i];
}

// x_proj_w (4,70,192) -> padded f16 (4,80,192) with zero rows 70..79
__global__ void cvt_xproj_kernel(const float* __restrict__ src, _Float16* __restrict__ dst){
  size_t n = (size_t)KD*CPJP*DIN;
  size_t i = (size_t)blockIdx.x*blockDim.x + threadIdx.x;
  size_t stride = (size_t)gridDim.x*blockDim.x;
  for (; i < n; i += stride){
    int d = i % DIN; int c = (i / DIN) % CPJP; int k = i / (DIN*CPJP);
    dst[i] = (c < CPJ) ? (_Float16)src[((size_t)k*CPJ + c)*DIN + d] : (_Float16)0.f;
  }
}

// ------- GEMM 1: xz = x @ in_proj_w^T (NT x 384, K=96) -------
// wave = (m-tile, group of 4 n-tiles): A frag loaded once per k-step, reused 4x
__global__ void gemm_inproj_kernel(const _Float16* __restrict__ A, const _Float16* __restrict__ W,
                                   float* __restrict__ C){
  int wv   = (int)((blockIdx.x*(size_t)blockDim.x + threadIdx.x) >> 5);
  int lane = threadIdx.x & 31;
  const int MT = NT/16, NG = (2*DIN)/64;            // 1568 m-tiles x 6 n-groups
  if (wv >= MT*NG) return;
  int mt = wv / NG, g = wv % NG;
  int half = lane >> 4;
  int mrow = mt*16 + (lane & 15);
  int n0   = g*64  + (lane & 15);
  v8f acc[4] = {};
#pragma unroll
  for (int kk = 0; kk < 3; ++kk){
    v16h af = load_afrag(A + (size_t)mrow*DM + kk*32, half);
#pragma unroll
    for (int j = 0; j < 4; ++j){
      v16h bf = load_bfrag(W + (size_t)(n0 + j*16)*DM + kk*32 + 16*half);
      acc[j] = __builtin_amdgcn_wmma_f32_16x16x32_f16(false, af, false, bf, (short)0, acc[j], false, false);
    }
  }
#pragma unroll
  for (int j = 0; j < 4; ++j){
    float* crow = C + (size_t)(mt*16 + 8*half)*(2*DIN) + g*64 + j*16 + (lane & 15);
#pragma unroll
    for (int i = 0; i < 8; ++i) crow[(size_t)i*(2*DIN)] = acc[j][i];
  }
}

// ---------------- depthwise conv 3x3 + bias + silu -> f16 ----------------
__global__ void conv_kernel(const float* __restrict__ xz, const float* __restrict__ cw,
                            const float* __restrict__ cb, _Float16* __restrict__ xib){
  size_t i = (size_t)blockIdx.x*blockDim.x + threadIdx.x;
  if (i >= (size_t)NT*DIN) return;
  int c = (int)(i % DIN);
  size_t tkn = i / DIN;
  int b = (int)(tkn / LL); int l = (int)(tkn % LL);
  int y = l / WW, x = l % WW;
  float acc = cb[c];
#pragma unroll
  for (int dy = -1; dy <= 1; ++dy){
    int yy = y + dy; if (yy < 0 || yy >= HH) continue;
#pragma unroll
    for (int dx = -1; dx <= 1; ++dx){
      int xx = x + dx; if (xx < 0 || xx >= WW) continue;
      float v = xz[(((size_t)b*LL + yy*WW + xx)*(2*DIN)) + c];   // xi = channels [0,192)
      acc += v * cw[((dy+1)*3 + (dx+1))*DIN + c];
    }
  }
  xib[i] = (_Float16)siluf(acc);
}

// ------- GEMM 2: x_dbl[k] = xs_k @ x_proj_w[k]^T, per (k,b): 3136 x 80, K=192 -------
// wave = one m-tile, all 5 n-tiles: A frag reused 5x per k-step, 30 WMMAs/wave
__global__ void gemm_xproj_kernel(const _Float16* __restrict__ XIB, const _Float16* __restrict__ WX,
                                  float* __restrict__ XD){
  int wv   = (int)((blockIdx.x*(size_t)blockDim.x + threadIdx.x) >> 5);
  int lane = threadIdx.x & 31;
  const int MT = LL/16;                             // 196 m-tiles per (k,b)
  if (wv >= KD*BB*MT) return;
  int mt = wv % MT;
  int kb = wv / MT;
  int b  = kb % BB;
  int k  = kb / BB;
  int half = lane >> 4;
  int mloc = mt*16 + (lane & 15);
  int src  = permk(k, mloc);                        // cross-scan via row permutation
  const _Float16* arow = XIB + ((size_t)b*LL + src)*DIN;
  const _Float16* wbase = WX + (size_t)k*CPJP*DIN + (size_t)(lane & 15)*DIN + 16*half;
  v8f acc[5] = {};
#pragma unroll
  for (int kk = 0; kk < 6; ++kk){
    v16h af = load_afrag(arow + kk*32, half);
#pragma unroll
    for (int j = 0; j < 5; ++j){
      v16h bf = load_bfrag(wbase + (size_t)j*16*DIN + kk*32);
      acc[j] = __builtin_amdgcn_wmma_f32_16x16x32_f16(false, af, false, bf, (short)0, acc[j], false, false);
    }
  }
#pragma unroll
  for (int j = 0; j < 5; ++j){
    float* crow = XD + ((size_t)kb*LL + mt*16 + 8*half)*CPJP + j*16 + (lane & 15);
#pragma unroll
    for (int i = 0; i < 8; ++i) crow[(size_t)i*CPJP] = acc[j][i];
  }
}

// ---------------- dt / dA precompute (packed float2 for one b64 broadcast load) ----------------
__global__ void dtda_kernel(const float* __restrict__ XD, const float* __restrict__ dt_bias,
                            const float* __restrict__ A_logs, float2* __restrict__ DTDA){
  size_t n = (size_t)KD*BB*LL*RK;
  size_t i = (size_t)blockIdx.x*blockDim.x + threadIdx.x;
  if (i >= n) return;
  int r = (int)(i % RK);
  size_t kbl = i / RK;                // (k*8+b)*L + l
  int k  = (int)(kbl / LL) / BB;
  float v  = XD[kbl*CPJP + r];
  float dt = softplusf(v + dt_bias[k*RK + r]);
  float Av = -expf(A_logs[k*RK + r]);
  DTDA[i] = make_float2(dt, expf(dt * Av));
}

// ---------------- selective scan: wave per (b,k,r,d), lanes = state n ----------------
// butterfly reduction leaves y on all lanes; stash per-lane, coalesced b32/lane store per 32 steps
__global__ void scan_kernel(const float* __restrict__ XD, const float2* __restrict__ DTDA,
                            const _Float16* __restrict__ XIB, const float* __restrict__ Dsw,
                            float* __restrict__ YS){
  int wv   = (int)((blockIdx.x*(size_t)blockDim.x + threadIdx.x) >> 5);
  int lane = threadIdx.x & 31;
  if (wv >= BB*KD*RK*HD) return;
  int d = wv & 31;
  int r = (wv >> 5) % RK;
  int k = (wv / (32*RK)) & 3;
  int b = wv / (32*RK*KD);
  int c = r*HD + d;
  int kb = k*BB + b;
  float Dv = Dsw[(k*RK + r)*HD + d];
  const float*  xrow = XD   + (size_t)kb*LL*CPJP;
  const float2* drow = DTDA + (size_t)kb*LL*RK + r;
  float* yrow = YS + ((size_t)kb*DIN + c)*LL;       // [kb][c][L]
  float s = 0.f, ystash = 0.f;
  for (int l = 0; l < LL; ++l){
    const float* xd = xrow + (size_t)l*CPJP;
    float Bv = xd[RK + lane];                       // B_t[n]
    float Cv = xd[RK + DST + lane];                 // C_t[n]
    float2 da = drow[(size_t)l*RK];                 // (dt, dA) broadcast b64
    int p = permk(k, l);
    float xt = (float)XIB[((size_t)b*LL + p)*DIN + c];
    s = s*da.y + xt*da.x*Bv;
    float ysum = s*Cv;
#pragma unroll
    for (int off = 16; off; off >>= 1) ysum += __shfl_xor(ysum, off, 32);
    ysum += xt*Dv;
    if ((l & 31) == lane) ystash = ysum;
    if ((l & 31) == 31)   yrow[(l & ~31) + lane] = ystash;   // coalesced 128B wave store
  }
}

// ---------------- cross-merge + layernorm + gelu + silu(z) -> f16 ----------------
__global__ void merge_kernel(const float* __restrict__ YS, const float* __restrict__ xz,
                             const float* __restrict__ ln_g, const float* __restrict__ ln_b,
                             _Float16* __restrict__ YB){
  __shared__ float red[256];
  int t = blockIdx.x;                 // token id
  int c = threadIdx.x;                // channel 0..191
  int b = t / LL, l = t % LL;
  int hi = l / WW, wi = l % WW;
  int q1 = wi*HH + hi;
  float v = YS[((size_t)(0*BB + b)*DIN + c)*LL + l        ]
          + YS[((size_t)(1*BB + b)*DIN + c)*LL + q1       ]
          + YS[((size_t)(2*BB + b)*DIN + c)*LL + (LL-1-l )]
          + YS[((size_t)(3*BB + b)*DIN + c)*LL + (LL-1-q1)];
  red[c] = v; if (c < 64) red[DIN + c] = 0.f;
  __syncthreads();
  for (int s = 128; s > 0; s >>= 1){ if (c < s) red[c] += red[c + s]; __syncthreads(); }
  float mu = red[0] * (1.f/DIN);
  __syncthreads();
  float dv = v - mu;
  red[c] = dv*dv; if (c < 64) red[DIN + c] = 0.f;
  __syncthreads();
  for (int s = 128; s > 0; s >>= 1){ if (c < s) red[c] += red[c + s]; __syncthreads(); }
  float var = red[0] * (1.f/DIN);
  float yv = dv * rsqrtf(var + 1e-5f) * ln_g[c] + ln_b[c];
  yv = 0.5f*yv*(1.f + erff(yv*0.70710678118654752f));        // exact gelu
  float z = xz[(size_t)t*(2*DIN) + DIN + c];
  YB[(size_t)t*DIN + c] = (_Float16)(yv * siluf(z));
}

// ------- GEMM 3: out = yb @ out_proj_w^T (NT x 96, K=192) -------
// wave = one m-tile, all 6 n-tiles: 36 WMMAs/wave
__global__ void gemm_outproj_kernel(const _Float16* __restrict__ A, const _Float16* __restrict__ W,
                                    float* __restrict__ C){
  int wv   = (int)((blockIdx.x*(size_t)blockDim.x + threadIdx.x) >> 5);
  int lane = threadIdx.x & 31;
  const int MT = NT/16;                             // 1568 m-tiles
  if (wv >= MT) return;
  int mt = wv;
  int half = lane >> 4;
  const _Float16* arow  = A + (size_t)(mt*16 + (lane & 15))*DIN;
  const _Float16* wbase = W + (size_t)(lane & 15)*DIN + 16*half;
  v8f acc[6] = {};
#pragma unroll
  for (int kk = 0; kk < 6; ++kk){
    v16h af = load_afrag(arow + kk*32, half);
#pragma unroll
    for (int j = 0; j < 6; ++j){
      v16h bf = load_bfrag(wbase + (size_t)j*16*DIN + kk*32);
      acc[j] = __builtin_amdgcn_wmma_f32_16x16x32_f16(false, af, false, bf, (short)0, acc[j], false, false);
    }
  }
#pragma unroll
  for (int j = 0; j < 6; ++j){
    float* crow = C + (size_t)(mt*16 + 8*half)*DM + j*16 + (lane & 15);
#pragma unroll
    for (int i = 0; i < 8; ++i) crow[(size_t)i*DM] = acc[j][i];
  }
}

// ---------------- launcher ----------------
extern "C" void kernel_launch(void* const* d_in, const int* in_sizes, int n_in,
                              void* d_out, int out_size, void* d_ws, size_t ws_size,
                              hipStream_t stream){
  const float* x         = (const float*)d_in[0];
  const float* in_proj_w = (const float*)d_in[1];
  const float* conv_w    = (const float*)d_in[2];
  const float* conv_b    = (const float*)d_in[3];
  const float* x_proj_w  = (const float*)d_in[4];
  const float* Ds        = (const float*)d_in[5];
  const float* A_logs    = (const float*)d_in[6];
  const float* dt_bias   = (const float*)d_in[7];
  const float* ln_g      = (const float*)d_in[8];
  const float* ln_b      = (const float*)d_in[9];
  const float* out_proj_w= (const float*)d_in[10];

  char* ws = (char*)d_ws;
  _Float16* XB  = (_Float16*)(ws + OFF_XB);
  _Float16* WI  = (_Float16*)(ws + OFF_WI);
  _Float16* WX  = (_Float16*)(ws + OFF_WX);
  _Float16* WO  = (_Float16*)(ws + OFF_WO);
  float*    XZ  = (float*)   (ws + OFF_XZ);
  _Float16* XIB = (_Float16*)(ws + OFF_XIB);
  float*    XD  = (float*)   (ws + OFF_XDBL);
  float2*   DTA = (float2*)  (ws + OFF_DTDA);
  float*    YS  = (float*)   (ws + OFF_YS);
  _Float16* YB  = (_Float16*)(ws + OFF_YB);

  // 1) f16 conversions
  cvt_f16_kernel<<<2048, 256, 0, stream>>>(x, XB, (size_t)NT*DM);
  cvt_f16_kernel<<<64,   256, 0, stream>>>(in_proj_w, WI, (size_t)2*DIN*DM);
  cvt_f16_kernel<<<64,   256, 0, stream>>>(out_proj_w, WO, (size_t)DM*DIN);
  cvt_xproj_kernel<<<64, 256, 0, stream>>>(x_proj_w, WX);

  // 2) in_proj GEMM (WMMA): 1568 m-tiles x 6 n-groups, 8 waves/block
  gemm_inproj_kernel<<<((NT/16)*6)/8, 256, 0, stream>>>(XB, WI, XZ);

  // 3) depthwise conv + silu
  conv_kernel<<<(int)(((size_t)NT*DIN + 255)/256), 256, 0, stream>>>(XZ, conv_w, conv_b, XIB);

  // 4) x_proj GEMM (WMMA, permuted rows): 4*8*196 waves
  gemm_xproj_kernel<<<(KD*BB*(LL/16))/8, 256, 0, stream>>>(XIB, WX, XD);

  // 5) dt / dA
  dtda_kernel<<<(int)(((size_t)KD*BB*LL*RK + 255)/256), 256, 0, stream>>>(XD, dt_bias, A_logs, DTA);

  // 6) selective scan: 6144 waves
  scan_kernel<<<(BB*KD*RK*HD)/8, 256, 0, stream>>>(XD, DTA, XIB, Ds, YS);

  // 7) merge + LN + gelu + silu(z)
  merge_kernel<<<NT, DIN, 0, stream>>>(YS, XZ, ln_g, ln_b, YB);

  // 8) out_proj GEMM (WMMA): 1568 waves, 36 WMMAs each
  gemm_outproj_kernel<<<(NT/16)/8, 256, 0, stream>>>(YB, WO, (float*)d_out);
}
